// MVTEModel_84756884619965
// MI455X (gfx1250) — matrix operand order repaired and checked by reading, not compiled, not run on hardware
//
// MVTE hypergraph pipeline for MI455X (gfx1250, wave32, WMMA).
//
// Pipeline:
//   1) scatter-mean(entity_emb[200000,128]) over 1.5M et edges -> tri_agg[500000,128]
//   2) tri = gelu(tri_agg@g2_w1^T+b1)@g2_w2^T+b2            (fused WMMA bf16 MLP)
//   3) scatter-mean(tri[500000,256]) over 0.8M tt edges -> tet_agg[200000,256]
//   4) tet_proj = gelu(tet_agg@g3_w1^T+b1)@Wc^T+bc, Wc=te_w@g3_w2 (folded GEMM)
//   5) scatter-mean(tet_proj) over 0.8M ett edges -> v_topo accum [200000,128]
//   6) per-triple: count-normalize + LayerNorm + softmax-fusion + TransE score
//
// GEMM strategy: compute transposed tiles D' = Wtile(A) x Xtile(B).
//  - B-fragments (activations) are loop-invariant over output tiles: held in
//    VGPRs, loaded once per 16-row tile (single contiguous 32B read/lane).
//  - D' layout gives each lane 8 *consecutive* output columns -> epilogue
//    packs bf16 and stores one b128 per lane (no scattered b16 stores).
//  - nt loops kept rolled (#pragma unroll 1) to bound VGPR pressure.
//  - gelu uses a branchless A&S-7.1.26 erf (rcp + hw exp + 6 fma), replacing
//    libm erff whose exec-masked two-path expansion serialized the epilogue.
// Workspace peak ~745 MB (regions reused in stream order).

#include <hip/hip_runtime.h>
#include <hip/hip_bf16.h>
#include <math.h>

typedef __bf16 bf16_t;
typedef __attribute__((ext_vector_type(16))) __bf16 v16bf;
typedef __attribute__((ext_vector_type(8)))  float  v8f;

#define D_EMB   128
#define TRI_H   256
#define TET_H   256
#define NTRI_C  500000
#define NTET_C  200000

// ---------------------------------------------------------------------------
// Branchless exact-gelu: 0.5*x*(1+erf(x/sqrt2)) with A&S 7.1.26 erf
// (|err| <= 1.5e-7 — exact at bf16 output precision). ~15 straight-line VALU
// ops: v_rcp_f32, v_exp_f32, fmas, bfi(copysign). No EXEC divergence.
// ---------------------------------------------------------------------------
__device__ inline float gelu_exact(float x) {
    float z = fabsf(x) * 0.70710678118654752440f;   // |x|/sqrt(2)
    float t = __builtin_amdgcn_rcpf(fmaf(0.3275911f, z, 1.0f));
    float p = fmaf(t, 1.061405429f, -1.453152027f);
    p = fmaf(t, p, 1.421413741f);
    p = fmaf(t, p, -0.284496736f);
    p = fmaf(t, p, 0.254829592f);
    float e    = __expf(-z * z);                    // hw v_exp_f32
    float erfz = copysignf(1.0f - p * t * e, x);    // erf(x/sqrt2)
    return 0.5f * x * (1.0f + erfz);
}

// ---------------------------------------------------------------------------
// f32 -> bf16 elementwise (weight prep)
// ---------------------------------------------------------------------------
__global__ void f32_to_bf16_kernel(const float* __restrict__ in,
                                   bf16_t* __restrict__ out, int n) {
    for (int i = blockIdx.x * blockDim.x + threadIdx.x; i < n;
         i += gridDim.x * blockDim.x)
        out[i] = (bf16_t)in[i];
}

// ---------------------------------------------------------------------------
// Wc[n,k] = sum_t te_w[n,t]*g3_w2[t,k];  bc[n] = te_b[n] + sum_t te_w[n,t]*g3_b2[t]
// ---------------------------------------------------------------------------
__global__ void build_combined_w(const float* __restrict__ te_w,
                                 const float* __restrict__ g3_w2,
                                 const float* __restrict__ g3_b2,
                                 const float* __restrict__ te_b,
                                 bf16_t* __restrict__ wc,
                                 float* __restrict__ bc) {
    int idx = blockIdx.x * blockDim.x + threadIdx.x;
    if (idx >= D_EMB * TET_H) return;
    int n = idx >> 8;     // 0..127
    int k = idx & 255;    // 0..255
    float s = 0.f;
    for (int t = 0; t < TET_H; ++t)
        s += te_w[n * TET_H + t] * g3_w2[t * TET_H + k];
    wc[n * TET_H + k] = (bf16_t)s;
    if (k == 0) {
        float b = te_b[n];
        for (int t = 0; t < TET_H; ++t) b += te_w[n * TET_H + t] * g3_b2[t];
        bc[n] = b;
    }
}

// ---------------------------------------------------------------------------
// Scatter-add: one wave per edge. f32 source, D=128 (4 floats/lane).
// ---------------------------------------------------------------------------
__global__ __launch_bounds__(256)
void scatter_f32_128(const int* __restrict__ src_idx, const int* __restrict__ dst_idx,
                     const float* __restrict__ src, float* __restrict__ acc,
                     float* __restrict__ cnt, int nE) {
    int e    = (blockIdx.x * blockDim.x + threadIdx.x) >> 5;
    int lane = threadIdx.x & 31;
    if (e >= nE) return;
    int s = src_idx[e];
    int d = dst_idx[e];
    const float4 v = *(const float4*)(src + (size_t)s * 128 + lane * 4);
    float* out = acc + (size_t)d * 128 + lane * 4;
    atomicAdd(out + 0, v.x);
    atomicAdd(out + 1, v.y);
    atomicAdd(out + 2, v.z);
    atomicAdd(out + 3, v.w);
    if (lane == 0) atomicAdd(cnt + d, 1.0f);
}

// bf16 source, D=256 (8 elems/lane)
__global__ __launch_bounds__(256)
void scatter_bf16_256(const int* __restrict__ src_idx, const int* __restrict__ dst_idx,
                      const bf16_t* __restrict__ src, float* __restrict__ acc,
                      float* __restrict__ cnt, int nE) {
    int e    = (blockIdx.x * blockDim.x + threadIdx.x) >> 5;
    int lane = threadIdx.x & 31;
    if (e >= nE) return;
    int s = src_idx[e];
    int d = dst_idx[e];
    union { uint4 u; bf16_t h[8]; } t;
    t.u = *(const uint4*)(src + (size_t)s * 256 + lane * 8);
    float* out = acc + (size_t)d * 256 + lane * 8;
#pragma unroll
    for (int i = 0; i < 8; ++i) atomicAdd(out + i, (float)t.h[i]);
    if (lane == 0) atomicAdd(cnt + d, 1.0f);
}

// bf16 source, D=128 (4 elems/lane)
__global__ __launch_bounds__(256)
void scatter_bf16_128(const int* __restrict__ src_idx, const int* __restrict__ dst_idx,
                      const bf16_t* __restrict__ src, float* __restrict__ acc,
                      float* __restrict__ cnt, int nE) {
    int e    = (blockIdx.x * blockDim.x + threadIdx.x) >> 5;
    int lane = threadIdx.x & 31;
    if (e >= nE) return;
    int s = src_idx[e];
    int d = dst_idx[e];
    union { uint2 u; bf16_t h[4]; } t;
    t.u = *(const uint2*)(src + (size_t)s * 128 + lane * 4);
    float* out = acc + (size_t)d * 128 + lane * 4;
#pragma unroll
    for (int i = 0; i < 4; ++i) atomicAdd(out + i, (float)t.h[i]);
    if (lane == 0) atomicAdd(cnt + d, 1.0f);
}

// ---------------------------------------------------------------------------
// (acc, cnt) -> bf16 mean rows. logD = log2(D).
// ---------------------------------------------------------------------------
__global__ void normalize_to_bf16(const float* __restrict__ acc,
                                  const float* __restrict__ cnt,
                                  bf16_t* __restrict__ out, int rows, int logD) {
    size_t total = (size_t)rows << logD;
    for (size_t i = (size_t)blockIdx.x * blockDim.x + threadIdx.x; i < total;
         i += (size_t)gridDim.x * blockDim.x) {
        size_t r = i >> logD;
        float c  = cnt[r];
        float v  = (c > 0.0f) ? acc[i] / c : 0.0f;
        out[i]   = (bf16_t)v;
    }
}

// ---------------------------------------------------------------------------
// Fused 2-layer MLP with WMMA bf16 (transposed-tile formulation):
//   Y[rows,DOUT] = gelu(X[rows,DIN] @ W1[HID,DIN]^T + B1) @ W2[DOUT,HID]^T + B2
//
// One wave owns one 16-row tile. Per 16x16 output tile we compute
//   D'[n,m] = sum_k W[n,k] * Act[m,k]       (A = W rows, B = activations)
// Fragment layouts used here (same contiguous k order for A and B):
//   A (16x32, 16-bit): lane m=L&15, hi=L>>4; elem j -> K = hi*16 + j  (contig 32B)
//   B (32x16, 16-bit): lane m=L&15, hi=L>>4; elem j -> K = hi*16 + j  (contig 32B)
//   D: VGPR r, lane L -> row n = r + hi*8 (W row), col = L&15 (activation row)
// So each lane's 8 accumulators are CONSECUTIVE output columns -> pack to
// bf16 and store one b128. rows must be a multiple of 16 (500000/200000 are).
// ---------------------------------------------------------------------------
template <int DIN, int HID, int DOUT>
__global__ __launch_bounds__(128)
void fused_mlp_wmma(const bf16_t* __restrict__ X,
                    const bf16_t* __restrict__ W1, const float* __restrict__ B1,
                    const bf16_t* __restrict__ W2, const float* __restrict__ B2,
                    bf16_t* __restrict__ Y, int rows) {
    __shared__ bf16_t Hs[4][16 * HID];   // 8KB per wave (HID=256) -> 32KB/block
    const int lane = threadIdx.x & 31;
    const int wave = threadIdx.x >> 5;
    const int m    = lane & 15;          // activation row within tile / W row within tile
    const int hi   = lane >> 4;
    bf16_t* hrow = &Hs[wave][0];

    const int nRowTiles = rows >> 4;
    for (int tile = blockIdx.x * 4 + wave; tile < nRowTiles; tile += gridDim.x * 4) {
        const bf16_t* xrow = X + (size_t)(tile * 16 + m) * DIN;
        // ---- B-fragments of X: loop-invariant, held in registers ----
        v16bf bx[DIN / 32];
#pragma unroll
        for (int kc = 0; kc < DIN / 32; ++kc) {
            union { v16bf v; uint4 u[2]; } t;
            const uint4* p = (const uint4*)(xrow + kc * 32 + hi * 16);
            t.u[0] = p[0];
            t.u[1] = p[1];
            bx[kc] = t.v;
        }
        // ---- phase 1: H = gelu(X @ W1^T + B1) -> LDS (bf16, row-major) ----
#pragma unroll 1
        for (int nt = 0; nt < HID / 16; ++nt) {
            v8f acc = {};
            const bf16_t* wrow = W1 + (size_t)(nt * 16 + m) * DIN;
#pragma unroll
            for (int kc = 0; kc < DIN / 32; ++kc) {
                union { v16bf v; uint4 u[2]; } a;
                a.u[0] = *(const uint4*)(wrow + kc * 32 + hi * 16);
                a.u[1] = *(const uint4*)(wrow + kc * 32 + hi * 16 + 8);
                acc = __builtin_amdgcn_wmma_f32_16x16x32_bf16(
                    false, a.v, false, bx[kc], (short)0, acc, false, false);
            }
            // columns n = nt*16 + hi*8 + r (consecutive): bias, gelu, pack, one store
            const float4 b0 = *(const float4*)(B1 + nt * 16 + hi * 8);
            const float4 b1 = *(const float4*)(B1 + nt * 16 + hi * 8 + 4);
            const float bias[8] = {b0.x, b0.y, b0.z, b0.w, b1.x, b1.y, b1.z, b1.w};
            union { uint4 u; bf16_t h[8]; } pk;
#pragma unroll
            for (int r = 0; r < 8; ++r)
                pk.h[r] = (bf16_t)gelu_exact(acc[r] + bias[r]);
            *(uint4*)(hrow + (size_t)m * HID + nt * 16 + hi * 8) = pk.u;
        }
        // ---- B-fragments of H: loaded once, reused across all output tiles ----
        v16bf bh[HID / 32];
#pragma unroll
        for (int kc = 0; kc < HID / 32; ++kc) {
            union { v16bf v; uint4 u[2]; } t;
            const uint4* p = (const uint4*)(hrow + (size_t)m * HID + kc * 32 + hi * 16);
            t.u[0] = p[0];
            t.u[1] = p[1];
            bh[kc] = t.v;
        }
        // ---- phase 2: Y = H @ W2^T + B2 ----
#pragma unroll 1
        for (int nt = 0; nt < DOUT / 16; ++nt) {
            v8f acc = {};
            const bf16_t* wrow = W2 + (size_t)(nt * 16 + m) * HID;
#pragma unroll
            for (int kc = 0; kc < HID / 32; ++kc) {
                union { v16bf v; uint4 u[2]; } a;
                a.u[0] = *(const uint4*)(wrow + kc * 32 + hi * 16);
                a.u[1] = *(const uint4*)(wrow + kc * 32 + hi * 16 + 8);
                acc = __builtin_amdgcn_wmma_f32_16x16x32_bf16(
                    false, a.v, false, bh[kc], (short)0, acc, false, false);
            }
            const float4 b0 = *(const float4*)(B2 + nt * 16 + hi * 8);
            const float4 b1 = *(const float4*)(B2 + nt * 16 + hi * 8 + 4);
            const float bias[8] = {b0.x, b0.y, b0.z, b0.w, b1.x, b1.y, b1.z, b1.w};
            union { uint4 u; bf16_t h[8]; } pk;
#pragma unroll
            for (int r = 0; r < 8; ++r)
                pk.h[r] = (bf16_t)(acc[r] + bias[r]);
            *(uint4*)(Y + (size_t)(tile * 16 + m) * DOUT + nt * 16 + hi * 8) = pk.u;
        }
    }
}

// ---------------------------------------------------------------------------
// Final scoring: one wave per triple. Lazily count-normalizes + LayerNorms
// only the v_topo rows actually referenced, fuses with entity_emb via
// softmax(fusion_alpha), computes gamma - ||V[h]+rel[r]-V[t]||.
// ---------------------------------------------------------------------------
__device__ inline float wave_sum(float v) {
#pragma unroll
    for (int off = 16; off >= 1; off >>= 1) v += __shfl_xor(v, off, 32);
    return v;
}

__device__ inline void topo_ln_row(const float* __restrict__ vacc,
                                   const float* __restrict__ vcnt,
                                   const float* __restrict__ lnw,
                                   const float* __restrict__ lnb,
                                   int row, int lane, float outv[4]) {
    float c   = vcnt[row];
    float inv = (c > 0.f) ? (1.f / c) : 0.f;   // cnt==0 -> zero row -> LN gives bias
    const float4 a = *(const float4*)(vacc + (size_t)row * 128 + lane * 4);
    float x[4] = {a.x * inv, a.y * inv, a.z * inv, a.w * inv};
    float mu = wave_sum(x[0] + x[1] + x[2] + x[3]) * (1.0f / 128.0f);
    float q  = 0.f;
#pragma unroll
    for (int i = 0; i < 4; ++i) { float d = x[i] - mu; q += d * d; }
    float var  = wave_sum(q) * (1.0f / 128.0f);
    float rstd = rsqrtf(var + 1e-5f);
#pragma unroll
    for (int i = 0; i < 4; ++i)
        outv[i] = (x[i] - mu) * rstd * lnw[lane * 4 + i] + lnb[lane * 4 + i];
}

__global__ __launch_bounds__(256)
void final_score(const float* __restrict__ ent, const float* __restrict__ rel,
                 const float* __restrict__ vacc, const float* __restrict__ vcnt,
                 const float* __restrict__ lnw, const float* __restrict__ lnb,
                 const float* __restrict__ fa, const float* __restrict__ gamma,
                 const int* __restrict__ triples, float* __restrict__ out, int B) {
    int t    = (blockIdx.x * blockDim.x + threadIdx.x) >> 5;
    int lane = threadIdx.x & 31;
    if (t >= B) return;
    int h  = triples[t * 3 + 0];
    int r  = triples[t * 3 + 1];
    int tl = triples[t * 3 + 2];

    float a0 = fa[0], a1 = fa[1];
    float mx = fmaxf(a0, a1);
    float e0 = expf(a0 - mx), e1 = expf(a1 - mx);
    float w0 = e0 / (e0 + e1), w1 = e1 / (e0 + e1);

    float lh[4], lt[4];
    topo_ln_row(vacc, vcnt, lnw, lnb, h, lane, lh);
    topo_ln_row(vacc, vcnt, lnw, lnb, tl, lane, lt);

    const float4 eh = *(const float4*)(ent + (size_t)h * 128 + lane * 4);
    const float4 et = *(const float4*)(ent + (size_t)tl * 128 + lane * 4);
    const float4 rr = *(const float4*)(rel + (size_t)r * 128 + lane * 4);
    float ehv[4] = {eh.x, eh.y, eh.z, eh.w};
    float etv[4] = {et.x, et.y, et.z, et.w};
    float rrv[4] = {rr.x, rr.y, rr.z, rr.w};

    float ss = 0.f;
#pragma unroll
    for (int i = 0; i < 4; ++i) {
        float Vh = w0 * ehv[i] + w1 * lh[i];
        float Vt = w0 * etv[i] + w1 * lt[i];
        float d  = Vh + rrv[i] - Vt;
        ss += d * d;
    }
    ss = wave_sum(ss);
    if (lane == 0) out[t] = gamma[0] - sqrtf(ss);
}

// ---------------------------------------------------------------------------
// Host-side orchestration (graph-capture safe: only kernel launches +
// hipMemsetAsync on `stream`).
// ---------------------------------------------------------------------------
extern "C" void kernel_launch(void* const* d_in, const int* in_sizes, int n_in,
                              void* d_out, int out_size, void* d_ws, size_t ws_size,
                              hipStream_t stream) {
    (void)n_in; (void)out_size; (void)ws_size;
    const float* entity_emb   = (const float*)d_in[0];
    const float* relation_emb = (const float*)d_in[1];
    const float* g2_w1 = (const float*)d_in[2];
    const float* g2_b1 = (const float*)d_in[3];
    const float* g2_w2 = (const float*)d_in[4];
    const float* g2_b2 = (const float*)d_in[5];
    const float* g3_w1 = (const float*)d_in[6];
    const float* g3_b1 = (const float*)d_in[7];
    const float* g3_w2 = (const float*)d_in[8];
    const float* g3_b2 = (const float*)d_in[9];
    const float* te_w  = (const float*)d_in[10];
    const float* te_b  = (const float*)d_in[11];
    const float* ln_w  = (const float*)d_in[12];
    const float* ln_b  = (const float*)d_in[13];
    const float* fusion_alpha = (const float*)d_in[14];
    const float* gamma = (const float*)d_in[15];
    const int* triples       = (const int*)d_in[16];
    const int* et_index      = (const int*)d_in[17];
    const int* tt_index      = (const int*)d_in[18];
    const int* ent_tet_index = (const int*)d_in[19];

    const int B_TR  = in_sizes[16] / 3;
    const int E_ET  = in_sizes[17] / 2;
    const int E_TT  = in_sizes[18] / 2;
    const int E_ETT = in_sizes[19] / 2;

    // ---- workspace regions (overlapped in stream order; peak ~745 MB) ----
    char* ws = (char*)d_ws;
    const size_t R0 = 0;                        // 256,000,000: tri_acc -> tet_acc
    const size_t R1 = 256000000;                // 128,000,000: x1_bf16 -> vt_acc+vt_cnt
    const size_t R2 = R1 + 128000000;           // 256,000,000: tri_bf16 -> tp_bf16
    const size_t R3 = R2 + 256000000;           // 102,400,000: x2_bf16
    const size_t R4 = R3 + 102400000;           //   2,097,152: shared count buffer
    const size_t R5 = R4 + 2097152;             // weights (~0.4 MB)

    float*  tri_acc = (float*)(ws + R0);
    float*  tet_acc = (float*)(ws + R0);
    bf16_t* x1      = (bf16_t*)(ws + R1);
    float*  vt_acc  = (float*)(ws + R1);
    float*  vt_cnt  = (float*)(ws + R1 + 102400000);
    bf16_t* tri_b   = (bf16_t*)(ws + R2);
    bf16_t* tp_b    = (bf16_t*)(ws + R2);
    bf16_t* x2      = (bf16_t*)(ws + R3);
    float*  cntbuf  = (float*)(ws + R4);
    bf16_t* w1a = (bf16_t*)(ws + R5);                           // g2_w1 [256,128]
    bf16_t* w2a = (bf16_t*)(ws + R5 + 65536);                   // g2_w2 [256,256]
    bf16_t* w1b = (bf16_t*)(ws + R5 + 65536 + 131072);          // g3_w1 [256,256]
    bf16_t* wc  = (bf16_t*)(ws + R5 + 65536 + 2 * 131072);      // [128,256]
    float*  bc  = (float*)(ws + R5 + 2 * 65536 + 2 * 131072);   // [128]

    // ---- weight prep ----
    f32_to_bf16_kernel<<<64, 256, 0, stream>>>(g2_w1, w1a, TRI_H * D_EMB);
    f32_to_bf16_kernel<<<64, 256, 0, stream>>>(g2_w2, w2a, TRI_H * TRI_H);
    f32_to_bf16_kernel<<<64, 256, 0, stream>>>(g3_w1, w1b, TET_H * TRI_H);
    build_combined_w<<<(D_EMB * TET_H + 255) / 256, 256, 0, stream>>>(
        te_w, g3_w2, g3_b2, te_b, wc, bc);

    // ---- level 1: entity -> triangle ----
    hipMemsetAsync(tri_acc, 0, (size_t)NTRI_C * D_EMB * 4, stream);
    hipMemsetAsync(cntbuf, 0, (size_t)NTRI_C * 4, stream);
    scatter_f32_128<<<(E_ET * 32 + 255) / 256, 256, 0, stream>>>(
        et_index, et_index + E_ET, entity_emb, tri_acc, cntbuf, E_ET);
    normalize_to_bf16<<<4096, 256, 0, stream>>>(tri_acc, cntbuf, x1, NTRI_C, 7);
    fused_mlp_wmma<128, 256, 256><<<2048, 128, 0, stream>>>(
        x1, w1a, g2_b1, w2a, g2_b2, tri_b, NTRI_C);

    // ---- level 2: triangle -> tetra ----
    hipMemsetAsync(tet_acc, 0, (size_t)NTET_C * TRI_H * 4, stream);
    hipMemsetAsync(cntbuf, 0, (size_t)NTET_C * 4, stream);
    scatter_bf16_256<<<(E_TT * 32 + 255) / 256, 256, 0, stream>>>(
        tt_index, tt_index + E_TT, tri_b, tet_acc, cntbuf, E_TT);
    normalize_to_bf16<<<4096, 256, 0, stream>>>(tet_acc, cntbuf, x2, NTET_C, 8);
    fused_mlp_wmma<256, 256, 128><<<2048, 128, 0, stream>>>(
        x2, w1b, g3_b1, wc, bc, tp_b, NTET_C);

    // ---- level 3: tetra -> entity (v_topo accumulation) ----
    hipMemsetAsync(vt_acc, 0, (size_t)NTET_C * D_EMB * 4, stream);
    hipMemsetAsync(vt_cnt, 0, (size_t)NTET_C * 4, stream);
    scatter_bf16_128<<<(E_ETT * 32 + 255) / 256, 256, 0, stream>>>(
        ent_tet_index + E_ETT, ent_tet_index, tp_b, vt_acc, vt_cnt, E_ETT);

    // ---- scoring head (lazy LN on referenced rows only) ----
    final_score<<<(B_TR * 32 + 255) / 256, 256, 0, stream>>>(
        entity_emb, relation_emb, vt_acc, vt_cnt, ln_w, ln_b,
        fusion_alpha, gamma, triples, (float*)d_out, B_TR);
}
